// Model_5566277616477
// MI455X (gfx1250) — compile-verified, matrix operand order checked by reading, not run on hardware
//
#include <hip/hip_runtime.h>

#define USERN   100000
#define ITEMN   200000
#define NNODE   (USERN + ITEMN)
#define LATD    64
#define HYPERN  128
#define FDIM    16
#define GLAYERS 2
#define NEDGE   3000000
#define LN_EPS  1e-5f

typedef __attribute__((ext_vector_type(16))) __bf16 v16bf;
typedef __attribute__((ext_vector_type(8)))  __bf16 v8bf;
typedef __attribute__((ext_vector_type(8)))  float  v8f;

__device__ __forceinline__ v8f wmma_bf16(v16bf a, v16bf b, v8f c) {
  return __builtin_amdgcn_wmma_f32_16x16x32_bf16(false, a, false, b, (short)0, c, false, false);
}

// ---- generic (strided) f32 fragment loader: rows by lane, ISA 16-bit layout.
// element(e): row = row0 + (lane&15), K = k0 + (e<8?e:e+8) + 8*(lane>>4)
__device__ __forceinline__ v16bf frag_f32(const float* p, int rs, int cs, int row0, int k0) {
  const int lane = threadIdx.x & 31;
  const int lh = lane >> 4, r = lane & 15;
  const float* base = p + (long)(row0 + r) * rs;
  v16bf a;
#pragma unroll
  for (int e = 0; e < 16; ++e) {
    const int k = k0 + (e < 8 ? e : e + 8) + 8 * lh;
    a[e] = (__bf16)base[(long)k * cs];
  }
  return a;
}

// ---- row-major f32 loader: each lane's K elements are two contiguous 8-float
// runs -> 4x global_load_b128 + pack-convert.
__device__ __forceinline__ v16bf frag_f32_rm(const float* p, int rs, int row0, int k0) {
  const int lane = threadIdx.x & 31;
  const int lh = lane >> 4, r = lane & 15;
  const float* base = p + (long)(row0 + r) * rs + k0 + 8 * lh;
  float4 q0 = *(const float4*)(base + 0);
  float4 q1 = *(const float4*)(base + 4);
  float4 q2 = *(const float4*)(base + 16);
  float4 q3 = *(const float4*)(base + 20);
  v16bf a;
  a[0] = (__bf16)q0.x;  a[1] = (__bf16)q0.y;  a[2] = (__bf16)q0.z;  a[3] = (__bf16)q0.w;
  a[4] = (__bf16)q1.x;  a[5] = (__bf16)q1.y;  a[6] = (__bf16)q1.z;  a[7] = (__bf16)q1.w;
  a[8] = (__bf16)q2.x;  a[9] = (__bf16)q2.y;  a[10] = (__bf16)q2.z; a[11] = (__bf16)q2.w;
  a[12] = (__bf16)q3.x; a[13] = (__bf16)q3.y; a[14] = (__bf16)q3.z; a[15] = (__bf16)q3.w;
  return a;
}

// ---- row-major bf16 (LDS) loader: 2x ds_load_b128 per fragment.
__device__ __forceinline__ v16bf frag_bf16_rm(const __bf16* p, int rs, int row0, int k0) {
  const int lane = threadIdx.x & 31;
  const int lh = lane >> 4, r = lane & 15;
  const __bf16* base = p + (long)(row0 + r) * rs + k0 + 8 * lh;
  v8bf lo = *(const v8bf*)(base);
  v8bf hi = *(const v8bf*)(base + 16);
  v16bf a;
#pragma unroll
  for (int e = 0; e < 8; ++e) { a[e] = lo[e]; a[e + 8] = hi[e]; }
  return a;
}

__device__ __forceinline__ float leaky(float v) { return v > 0.f ? v : 0.5f * v; }

// LayerNorm over the 64-feature row held in D-layout val[4 ntiles][8 rows].
__device__ __forceinline__ void ln_store(float val[4][8], float* out, int t0,
                                         const float* g, const float* b) {
  const int lane = threadIdx.x & 31;
  const int lh = lane >> 4, n = lane & 15;
  float mean[8], var[8];
#pragma unroll
  for (int r = 0; r < 8; ++r) {
    float s = val[0][r] + val[1][r] + val[2][r] + val[3][r];
#pragma unroll
    for (int m = 1; m <= 8; m <<= 1) s += __shfl_xor(s, m, 32);
    mean[r] = s * (1.f / 64.f);
  }
#pragma unroll
  for (int r = 0; r < 8; ++r) {
    float s = 0.f;
#pragma unroll
    for (int nt = 0; nt < 4; ++nt) { float d = val[nt][r] - mean[r]; s += d * d; }
#pragma unroll
    for (int m = 1; m <= 8; m <<= 1) s += __shfl_xor(s, m, 32);
    var[r] = s * (1.f / 64.f);
  }
#pragma unroll
  for (int nt = 0; nt < 4; ++nt) {
    const int col = nt * 16 + n;
    const float gg = g[col], bb = b[col];
#pragma unroll
    for (int r = 0; r < 8; ++r) {
      float y = (val[nt][r] - mean[r]) * rsqrtf(var[r] + LN_EPS) * gg + bb;
      out[(long)(t0 + r + 8 * lh) * LATD + col] = y;
    }
  }
}

// ---------------- weight prep: f32 [K,C] -> bf16 transposed [C,Kpad] ----------
__global__ void k_wprep(const float* __restrict__ src, __bf16* __restrict__ dst,
                        int K, int C, int Kpad) {
  int i = blockIdx.x * blockDim.x + threadIdx.x;
  if (i >= C * Kpad) return;
  int o = i / Kpad, k = i % Kpad;
  dst[i] = (k < K) ? (__bf16)src[(long)k * C + o] : (__bf16)0.f;
}

// ---------------- init: x = total = concat(uE, iE) ---------------------------
__global__ void k_init(const float* __restrict__ uE, const float* __restrict__ iE,
                       float* __restrict__ x, float* __restrict__ total, int nu, int ntot) {
  int i = blockIdx.x * blockDim.x + threadIdx.x;
  if (i >= ntot) return;
  float v = (i < nu) ? uE[i] : iE[i - nu];
  x[i] = v; total[i] = v;
}

// ---------------- uu = E @ Hyp  (WMMA) + attention logits --------------------
__global__ __launch_bounds__(256) void k_uuprep(const float* __restrict__ E,
                                                const float* __restrict__ Hyp,
                                                const float* __restrict__ attn,
                                                float* __restrict__ out,
                                                float* __restrict__ logit, int nrows) {
  // stage Hyp^T as bf16 in LDS: sHyp[h*64 + k] = Hyp[k*128 + h]
  __shared__ __align__(16) __bf16 sHyp[HYPERN * LATD];
  for (int i = threadIdx.x; i < HYPERN * LATD; i += blockDim.x) {
    int h = i >> 6, k = i & 63;
    sHyp[i] = (__bf16)Hyp[k * HYPERN + h];
  }
  __syncthreads();

  const int wave = threadIdx.x >> 5;
  const int lane = threadIdx.x & 31;
  const int lh = lane >> 4, n = lane & 15;
  const int t0 = (blockIdx.x * 8 + wave) * 16;
  if (t0 >= nrows) return;
  __builtin_prefetch(E + (long)(t0 + 128) * LATD, 0, 1);
  v16bf a0 = frag_f32_rm(E, LATD, t0, 0);
  v16bf a1 = frag_f32_rm(E, LATD, t0, 32);
  float lacc[8] = {0.f, 0.f, 0.f, 0.f, 0.f, 0.f, 0.f, 0.f};
#pragma unroll
  for (int ht = 0; ht < HYPERN / 16; ++ht) {
    v8f d = {};
    d = wmma_bf16(a0, frag_bf16_rm(sHyp, LATD, ht * 16, 0), d);
    d = wmma_bf16(a1, frag_bf16_rm(sHyp, LATD, ht * 16, 32), d);
    float av = attn[ht * 16 + n];
#pragma unroll
    for (int r = 0; r < 8; ++r) {
      out[(long)(t0 + r + 8 * lh) * HYPERN + ht * 16 + n] = d[r];
      lacc[r] += d[r] * av;
    }
  }
#pragma unroll
  for (int r = 0; r < 8; ++r) {
#pragma unroll
    for (int m = 1; m <= 8; m <<= 1) lacc[r] += __shfl_xor(lacc[r], m, 32);
  }
  if (n == 0) {
#pragma unroll
    for (int r = 0; r < 8; ++r) logit[t0 + 8 * lh + r] = lacc[r];
  }
}

// ---------------- softmax-over-nodes reductions ------------------------------
__global__ void k_rmax(const float* __restrict__ v, int n, float* __restrict__ part) {
  __shared__ float s[256];
  float m = -3.4e38f;
  for (int i = blockIdx.x * blockDim.x + threadIdx.x; i < n; i += gridDim.x * blockDim.x)
    m = fmaxf(m, v[i]);
  s[threadIdx.x] = m; __syncthreads();
  for (int o = 128; o; o >>= 1) {
    if ((int)threadIdx.x < o) s[threadIdx.x] = fmaxf(s[threadIdx.x], s[threadIdx.x + o]);
    __syncthreads();
  }
  if (!threadIdx.x) part[blockIdx.x] = s[0];
}

__global__ void k_rsum(const float* __restrict__ v, int n, const float* __restrict__ red,
                       int mslot, float* __restrict__ part) {
  __shared__ float s[256];
  float gm = red[mslot];
  float acc = 0.f;
  for (int i = blockIdx.x * blockDim.x + threadIdx.x; i < n; i += gridDim.x * blockDim.x)
    acc += __expf(v[i] - gm);
  s[threadIdx.x] = acc; __syncthreads();
  for (int o = 128; o; o >>= 1) {
    if ((int)threadIdx.x < o) s[threadIdx.x] += s[threadIdx.x + o];
    __syncthreads();
  }
  if (!threadIdx.x) part[blockIdx.x] = s[0];
}

__global__ void k_fin(const float* __restrict__ part, int nb, float* __restrict__ red,
                      int slot, int isMax) {
  __shared__ float s[256];
  float v = isMax ? -3.4e38f : 0.f;
  for (int i = threadIdx.x; i < nb; i += 256)
    v = isMax ? fmaxf(v, part[i]) : v + part[i];
  s[threadIdx.x] = v; __syncthreads();
  for (int o = 128; o; o >>= 1) {
    if ((int)threadIdx.x < o)
      s[threadIdx.x] = isMax ? fmaxf(s[threadIdx.x], s[threadIdx.x + o])
                             : s[threadIdx.x] + s[threadIdx.x + o];
    __syncthreads();
  }
  if (!threadIdx.x) red[slot] = s[0];
}

__global__ void k_scale_rows(float* __restrict__ mat, const float* __restrict__ logit,
                             int n, int H, const float* __restrict__ red, int mslot, int sslot) {
  int i = blockIdx.x * blockDim.x + threadIdx.x;
  if (i >= n * H) return;
  int row = i / H;
  mat[i] *= __expf(logit[row] - red[mslot]) / red[sslot];
}

// ---------------- SpMM: out[r] += val * x[c]  (COO, atomic scatter) ----------
__global__ void k_spmm(const float* __restrict__ vals, const int* __restrict__ rows,
                       const int* __restrict__ cols, const float* __restrict__ x,
                       float* __restrict__ out, int E) {
  int tid = blockIdx.x * blockDim.x + threadIdx.x;
  int e = tid >> 4;
  if (e >= E) return;
  int d4 = (tid & 15) * 4;
  float v = vals[e];
  long r = rows[e], c = cols[e];
  const float4 xv = *(const float4*)(x + c * LATD + d4);
  float* o = out + r * LATD + d4;
  atomicAdd(o + 0, v * xv.x);
  atomicAdd(o + 1, v * xv.y);
  atomicAdd(o + 2, v * xv.z);
  atomicAdd(o + 3, v * xv.w);
}

// ---------------- P = U^T @ X  (WMMA reduction over node chunks) -------------
#define PT_CHUNK 8192
__global__ __launch_bounds__(256) void k_pt(const float* __restrict__ U,
                                            const float* __restrict__ X,
                                            float* __restrict__ P, int nodes) {
  const int gwave = blockIdx.x * 8 + (threadIdx.x >> 5);
  const int ntp = (HYPERN / 16) * (LATD / 16);           // 32 tile pairs
  const int chunk = gwave / ntp;
  const int tp = gwave % ntp;
  const int ht = tp >> 2, ft = tp & 3;
  const int k0 = chunk * PT_CHUNK;
  if (k0 >= nodes) return;
  const int kend = min(nodes, k0 + PT_CHUNK);            // chunk ends are /32
  v8f acc = {};
  for (int k = k0; k < kend; k += 32)
    acc = wmma_bf16(frag_f32(U, 1, HYPERN, ht * 16, k),
                    frag_f32(X, 1, LATD,   ft * 16, k), acc);
  const int lane = threadIdx.x & 31;
  const int lh = lane >> 4, n = lane & 15;
#pragma unroll
  for (int r = 0; r < 8; ++r)
    atomicAdd(&P[(ht * 16 + r + 8 * lh) * LATD + ft * 16 + n], acc[r]);
}

// ---------------- out = U @ P  (tall-skinny WMMA GEMM) -----------------------
__global__ __launch_bounds__(256) void k_hyp(const float* __restrict__ U,
                                             const float* __restrict__ P,
                                             float* __restrict__ out, int nrows) {
  // stage P^T as bf16 in LDS: sP[f*128 + h] = P[h*64 + f]
  __shared__ __align__(16) __bf16 sP[LATD * HYPERN];
  for (int i = threadIdx.x; i < LATD * HYPERN; i += blockDim.x) {
    int f = i >> 7, h = i & 127;
    sP[i] = (__bf16)P[h * LATD + f];
  }
  __syncthreads();

  const int wave = threadIdx.x >> 5;
  const int lane = threadIdx.x & 31;
  const int lh = lane >> 4, n = lane & 15;
  const int t0 = (blockIdx.x * 8 + wave) * 16;
  if (t0 >= nrows) return;
  __builtin_prefetch(U + (long)(t0 + 128) * HYPERN, 0, 1);
  v16bf a[4];
#pragma unroll
  for (int kt = 0; kt < 4; ++kt) a[kt] = frag_f32_rm(U, HYPERN, t0, kt * 32);
#pragma unroll
  for (int ft = 0; ft < 4; ++ft) {
    v8f d = {};
#pragma unroll
    for (int kt = 0; kt < 4; ++kt)
      d = wmma_bf16(a[kt], frag_bf16_rm(sP, HYPERN, ft * 16, kt * 32), d);
#pragma unroll
    for (int r = 0; r < 8; ++r)
      out[(long)(t0 + r + 8 * lh) * LATD + ft * 16 + n] = d[r];
  }
}

// ---------------- b128 LDS staging helper ------------------------------------
__device__ __forceinline__ void stage16(const __bf16* __restrict__ g, __bf16* s, int nelem) {
  const uint4* gs = (const uint4*)g;
  uint4* ds = (uint4*)s;
  for (int i = threadIdx.x; i < (nelem >> 3); i += blockDim.x) ds[i] = gs[i];
}

// ---------------- fused disentangler (in-place on z) -------------------------
#define WPB 4
__global__ __launch_bounds__(128) void k_dis(
    float* __restrict__ z, int nrows,
    const __bf16* __restrict__ w1t, const __bf16* __restrict__ w2t,
    const __bf16* __restrict__ wr1t, const __bf16* __restrict__ wr2t,
    const float* __restrict__ b1, const float* __restrict__ b2,
    const float* __restrict__ br1, const float* __restrict__ br2,
    const float* __restrict__ wk, const float* __restrict__ lng,
    const float* __restrict__ lnb) {
  __shared__ __align__(16) __bf16 sW1[4 * 64 * 64];
  __shared__ __align__(16) __bf16 sW2[4 * 16 * 64];
  __shared__ __align__(16) __bf16 sR1[64 * 32];
  __shared__ __align__(16) __bf16 sR2[64 * 64];
  __shared__ __align__(16) __bf16 scr[WPB][16 * 64];

  stage16(w1t, sW1, 4 * 64 * 64);
  stage16(w2t, sW2, 4 * 16 * 64);
  stage16(wr1t, sR1, 64 * 32);
  stage16(wr2t, sR2, 64 * 64);
  __syncthreads();

  const int wave = threadIdx.x >> 5;
  const int lane = threadIdx.x & 31;
  const int lh = lane >> 4, n = lane & 15;
  const int t0 = (blockIdx.x * WPB + wave) * 16;
  if (t0 >= nrows) return;
  __bf16* ms = scr[wave];
  __builtin_prefetch(z + (long)(t0 + WPB * 16 * 32) * LATD, 0, 1);

  // softmax(dis_weights)
  float wv0 = wk[0], wv1 = wk[1], wv2 = wk[2], wv3 = wk[3];
  float mw = fmaxf(fmaxf(wv0, wv1), fmaxf(wv2, wv3));
  float e0 = __expf(wv0 - mw), e1 = __expf(wv1 - mw), e2 = __expf(wv2 - mw), e3 = __expf(wv3 - mw);
  float inv = 1.f / (e0 + e1 + e2 + e3);
  float sw[4] = {e0 * inv, e1 * inv, e2 * inv, e3 * inv};

  v16bf az[2];
  az[0] = frag_f32_rm(z, LATD, t0, 0);
  az[1] = frag_f32_rm(z, LATD, t0, 32);

  float zD[4][8];
#pragma unroll
  for (int nt = 0; nt < 4; ++nt)
#pragma unroll
    for (int r = 0; r < 8; ++r)
      zD[nt][r] = z[(long)(t0 + r + 8 * lh) * LATD + nt * 16 + n];

  float comb[8] = {0.f, 0.f, 0.f, 0.f, 0.f, 0.f, 0.f, 0.f};

  for (int k = 0; k < 4; ++k) {
    v8f h[4];
#pragma unroll
    for (int nt = 0; nt < 4; ++nt) {
      float bv = b1[k * 64 + nt * 16 + n];
#pragma unroll
      for (int r = 0; r < 8; ++r) h[nt][r] = bv;
#pragma unroll
      for (int kt = 0; kt < 2; ++kt)
        h[nt] = wmma_bf16(az[kt], frag_bf16_rm(sW1 + k * 64 * 64, 64, nt * 16, kt * 32), h[nt]);
    }
    // leaky, D-layout -> row-major bf16 scratch
#pragma unroll
    for (int nt = 0; nt < 4; ++nt)
#pragma unroll
      for (int r = 0; r < 8; ++r)
        ms[(r + 8 * lh) * 64 + nt * 16 + n] = (__bf16)leaky(h[nt][r]);
    v16bf ah0 = frag_bf16_rm(ms, 64, 0, 0);
    v16bf ah1 = frag_bf16_rm(ms, 64, 0, 32);
    v8f f;
    float b2v = b2[k * 16 + n];
#pragma unroll
    for (int r = 0; r < 8; ++r) f[r] = b2v;
    f = wmma_bf16(ah0, frag_bf16_rm(sW2 + k * 16 * 64, 64, 0, 0), f);
    f = wmma_bf16(ah1, frag_bf16_rm(sW2 + k * 16 * 64, 64, 0, 32), f);
#pragma unroll
    for (int r = 0; r < 8; ++r) comb[r] += sw[k] * f[r];
  }

  // comb (16x16) -> K=32 zero-padded scratch; rec1 = leaky(comb @ Wr1 + br1)
#pragma unroll
  for (int r = 0; r < 8; ++r) {
    ms[(r + 8 * lh) * 32 + n] = (__bf16)comb[r];
    ms[(r + 8 * lh) * 32 + 16 + n] = (__bf16)0.f;
  }
  v16bf acmb = frag_bf16_rm(ms, 32, 0, 0);
  v8f rec1[4];
#pragma unroll
  for (int nt = 0; nt < 4; ++nt) {
    float bv = br1[nt * 16 + n];
#pragma unroll
    for (int r = 0; r < 8; ++r) rec1[nt][r] = bv;
    rec1[nt] = wmma_bf16(acmb, frag_bf16_rm(sR1, 32, nt * 16, 0), rec1[nt]);
  }
#pragma unroll
  for (int nt = 0; nt < 4; ++nt)
#pragma unroll
    for (int r = 0; r < 8; ++r)
      ms[(r + 8 * lh) * 64 + nt * 16 + n] = (__bf16)leaky(rec1[nt][r]);
  v16bf ar0 = frag_bf16_rm(ms, 64, 0, 0);
  v16bf ar1 = frag_bf16_rm(ms, 64, 0, 32);

  float val[4][8];
#pragma unroll
  for (int nt = 0; nt < 4; ++nt) {
    v8f y;
    float bv = br2[nt * 16 + n];
#pragma unroll
    for (int r = 0; r < 8; ++r) y[r] = bv;
    y = wmma_bf16(ar0, frag_bf16_rm(sR2, 64, nt * 16, 0), y);
    y = wmma_bf16(ar1, frag_bf16_rm(sR2, 64, nt * 16, 32), y);
#pragma unroll
    for (int r = 0; r < 8; ++r) val[nt][r] = zD[nt][r] + y[r];
  }
  ln_store(val, z, t0, lng, lnb);
}

// ---------------- fused gate (in-place on cur) -------------------------------
__global__ __launch_bounds__(128) void k_gate(
    float* __restrict__ cur, const float* __restrict__ prev, int nrows,
    const __bf16* __restrict__ wg1t, const __bf16* __restrict__ wg2t,
    const __bf16* __restrict__ wr1t, const __bf16* __restrict__ wr2t,
    const float* __restrict__ bg1, const float* __restrict__ bg2,
    const float* __restrict__ br1, const float* __restrict__ br2,
    const float* __restrict__ lng, const float* __restrict__ lnb) {
  __shared__ __align__(16) __bf16 sG1[64 * 128];
  __shared__ __align__(16) __bf16 sG2[64 * 64];
  __shared__ __align__(16) __bf16 sR1[64 * 128];
  __shared__ __align__(16) __bf16 sR2[64 * 64];
  __shared__ __align__(16) __bf16 scr[WPB][16 * 64];

  stage16(wg1t, sG1, 64 * 128);
  stage16(wg2t, sG2, 64 * 64);
  stage16(wr1t, sR1, 64 * 128);
  stage16(wr2t, sR2, 64 * 64);
  __syncthreads();

  const int wave = threadIdx.x >> 5;
  const int lane = threadIdx.x & 31;
  const int lh = lane >> 4, n = lane & 15;
  const int t0 = (blockIdx.x * WPB + wave) * 16;
  if (t0 >= nrows) return;
  __bf16* ms = scr[wave];
  __builtin_prefetch(prev + (long)(t0 + WPB * 16 * 32) * LATD, 0, 1);

  v16bf ac[4];
  ac[0] = frag_f32_rm(cur, LATD, t0, 0);
  ac[1] = frag_f32_rm(cur, LATD, t0, 32);
  ac[2] = frag_f32_rm(prev, LATD, t0, 0);
  ac[3] = frag_f32_rm(prev, LATD, t0, 32);

  float curD[4][8];
#pragma unroll
  for (int nt = 0; nt < 4; ++nt)
#pragma unroll
    for (int r = 0; r < 8; ++r)
      curD[nt][r] = cur[(long)(t0 + r + 8 * lh) * LATD + nt * 16 + n];

  // g1 = relu(c @ Wg1 + bg1)
#pragma unroll
  for (int nt = 0; nt < 4; ++nt) {
    v8f g1;
    float bv = bg1[nt * 16 + n];
#pragma unroll
    for (int r = 0; r < 8; ++r) g1[r] = bv;
#pragma unroll
    for (int kt = 0; kt < 4; ++kt)
      g1 = wmma_bf16(ac[kt], frag_bf16_rm(sG1, 128, nt * 16, kt * 32), g1);
#pragma unroll
    for (int r = 0; r < 8; ++r)
      ms[(r + 8 * lh) * 64 + nt * 16 + n] = (__bf16)fmaxf(g1[r], 0.f);
  }
  v16bf ag0 = frag_bf16_rm(ms, 64, 0, 0);
  v16bf ag1 = frag_bf16_rm(ms, 64, 0, 32);

  // gw = sigmoid(g1 @ Wg2 + bg2)
  float gw[4][8];
#pragma unroll
  for (int nt = 0; nt < 4; ++nt) {
    v8f y;
    float bv = bg2[nt * 16 + n];
#pragma unroll
    for (int r = 0; r < 8; ++r) y[r] = bv;
    y = wmma_bf16(ag0, frag_bf16_rm(sG2, 64, nt * 16, 0), y);
    y = wmma_bf16(ag1, frag_bf16_rm(sG2, 64, nt * 16, 32), y);
#pragma unroll
    for (int r = 0; r < 8; ++r) gw[nt][r] = 1.f / (1.f + __expf(-y[r]));
  }

  // res = leaky(c @ Wr1 + br1) @ Wr2 + br2
#pragma unroll
  for (int nt = 0; nt < 4; ++nt) {
    v8f r1;
    float bv = br1[nt * 16 + n];
#pragma unroll
    for (int r = 0; r < 8; ++r) r1[r] = bv;
#pragma unroll
    for (int kt = 0; kt < 4; ++kt)
      r1 = wmma_bf16(ac[kt], frag_bf16_rm(sR1, 128, nt * 16, kt * 32), r1);
#pragma unroll
    for (int r = 0; r < 8; ++r)
      ms[(r + 8 * lh) * 64 + nt * 16 + n] = (__bf16)leaky(r1[r]);
  }
  v16bf at0 = frag_bf16_rm(ms, 64, 0, 0);
  v16bf at1 = frag_bf16_rm(ms, 64, 0, 32);

  float val[4][8];
#pragma unroll
  for (int nt = 0; nt < 4; ++nt) {
    v8f y;
    float bv = br2[nt * 16 + n];
#pragma unroll
    for (int r = 0; r < 8; ++r) y[r] = bv;
    y = wmma_bf16(at0, frag_bf16_rm(sR2, 64, nt * 16, 0), y);
    y = wmma_bf16(at1, frag_bf16_rm(sR2, 64, nt * 16, 32), y);
#pragma unroll
    for (int r = 0; r < 8; ++r) val[nt][r] = gw[nt][r] * y[r] + curD[nt][r];
  }
  ln_store(val, cur, t0, lng, lnb);
}

// ---------------- x = a + b; total += x --------------------------------------
__global__ void k_combine(float* __restrict__ x, float* __restrict__ total,
                          const float* __restrict__ a, const float* __restrict__ b, int n) {
  int i = blockIdx.x * blockDim.x + threadIdx.x;
  if (i >= n) return;
  float v = a[i] + b[i];
  x[i] = v;
  total[i] += v;
}

// =============================================================================
extern "C" void kernel_launch(void* const* d_in, const int* in_sizes, int n_in,
                              void* d_out, int out_size, void* d_ws, size_t ws_size,
                              hipStream_t stream) {
  (void)in_sizes; (void)n_in; (void)out_size; (void)ws_size;
  const float* uEmbeds = (const float*)d_in[0];
  const float* iEmbeds = (const float*)d_in[1];
  const float* uHyp    = (const float*)d_in[2];
  const float* iHyp    = (const float*)d_in[3];
  const float* u_attn  = (const float*)d_in[4];
  const float* i_attn  = (const float*)d_in[5];
  const float* dW1  = (const float*)d_in[6];
  const float* db1  = (const float*)d_in[7];
  const float* dW2  = (const float*)d_in[8];
  const float* db2  = (const float*)d_in[9];
  const float* dWr1 = (const float*)d_in[10];
  const float* dbr1 = (const float*)d_in[11];
  const float* dWr2 = (const float*)d_in[12];
  const float* dbr2 = (const float*)d_in[13];
  const float* dwts = (const float*)d_in[14];
  const float* dlng = (const float*)d_in[15];
  const float* dlnb = (const float*)d_in[16];
  const float* gWg1 = (const float*)d_in[17];
  const float* gbg1 = (const float*)d_in[18];
  const float* gWg2 = (const float*)d_in[19];
  const float* gbg2 = (const float*)d_in[20];
  const float* gWr1 = (const float*)d_in[21];
  const float* gbr1 = (const float*)d_in[22];
  const float* gWr2 = (const float*)d_in[23];
  const float* gbr2 = (const float*)d_in[24];
  const float* glng = (const float*)d_in[25];
  const float* glnb = (const float*)d_in[26];
  const float* adj_vals = (const float*)d_in[27];
  const int* adj_rows = (const int*)d_in[28];
  const int* adj_cols = (const int*)d_in[29];
  float* total = (float*)d_out;

  // workspace bump allocator
  size_t off = 0;
  char* base = (char*)d_ws;
  auto alloc = [&](size_t bytes) -> void* {
    void* p = base + off;
    off = (off + bytes + 255) & ~(size_t)255;
    return p;
  };
  float* x    = (float*)alloc((size_t)NNODE * LATD * 4);
  float* tem  = (float*)alloc((size_t)NNODE * LATD * 4);
  float* hyp  = (float*)alloc((size_t)NNODE * LATD * 4);
  float* uu   = (float*)alloc((size_t)USERN * HYPERN * 4);
  float* ii   = (float*)alloc((size_t)ITEMN * HYPERN * 4);
  float* logit = (float*)alloc((size_t)ITEMN * 4);
  float* part = (float*)alloc(1024 * 4);
  float* red  = (float*)alloc(8 * 4);
  float* P    = (float*)alloc(HYPERN * LATD * 4);
  __bf16* W1t  = (__bf16*)alloc((size_t)GLAYERS * 2 * 4 * 64 * 64 * 2);
  __bf16* W2t  = (__bf16*)alloc((size_t)GLAYERS * 2 * 4 * 16 * 64 * 2);
  __bf16* Wr1t = (__bf16*)alloc((size_t)GLAYERS * 2 * 64 * 32 * 2);
  __bf16* Wr2t = (__bf16*)alloc((size_t)GLAYERS * 2 * 64 * 64 * 2);
  __bf16* G1t  = (__bf16*)alloc((size_t)GLAYERS * 64 * 128 * 2);
  __bf16* G2t  = (__bf16*)alloc((size_t)GLAYERS * 64 * 64 * 2);
  __bf16* GR1t = (__bf16*)alloc((size_t)GLAYERS * 64 * 128 * 2);
  __bf16* GR2t = (__bf16*)alloc((size_t)GLAYERS * 64 * 64 * 2);

  auto wprep = [&](const float* src, __bf16* dst, int K, int C, int Kpad) {
    int tot = C * Kpad;
    k_wprep<<<(tot + 255) / 256, 256, 0, stream>>>(src, dst, K, C, Kpad);
  };
  for (int i = 0; i < GLAYERS; ++i) {
    for (int j = 0; j < 2; ++j) {
      int ij = i * 2 + j;
      for (int k = 0; k < 4; ++k) {
        wprep(dW1 + (size_t)(ij * 4 + k) * 64 * 64, W1t + (size_t)(ij * 4 + k) * 64 * 64, 64, 64, 64);
        wprep(dW2 + (size_t)(ij * 4 + k) * 64 * 16, W2t + (size_t)(ij * 4 + k) * 16 * 64, 64, 16, 64);
      }
      wprep(dWr1 + (size_t)ij * 16 * 64, Wr1t + (size_t)ij * 64 * 32, 16, 64, 32);
      wprep(dWr2 + (size_t)ij * 64 * 64, Wr2t + (size_t)ij * 64 * 64, 64, 64, 64);
    }
    wprep(gWg1 + (size_t)i * 128 * 64, G1t + (size_t)i * 64 * 128, 128, 64, 128);
    wprep(gWg2 + (size_t)i * 64 * 64,  G2t + (size_t)i * 64 * 64,  64, 64, 64);
    wprep(gWr1 + (size_t)i * 128 * 64, GR1t + (size_t)i * 64 * 128, 128, 64, 128);
    wprep(gWr2 + (size_t)i * 64 * 64,  GR2t + (size_t)i * 64 * 64,  64, 64, 64);
  }

  // x = total = concat(uE, iE)
  {
    int tot = NNODE * LATD;
    k_init<<<(tot + 255) / 256, 256, 0, stream>>>(uEmbeds, iEmbeds, x, total, USERN * LATD, tot);
  }

  // attention-weighted hyperedge incidence (softmax over node axis)
  auto attn_path = [&](const float* E, const float* Hy, const float* at, float* out,
                       int nodes, int mslot, int sslot) {
    int tiles = nodes / 16;
    k_uuprep<<<(tiles + 7) / 8, 256, 0, stream>>>(E, Hy, at, out, logit, nodes);
    k_rmax<<<256, 256, 0, stream>>>(logit, nodes, part);
    k_fin<<<1, 256, 0, stream>>>(part, 256, red, mslot, 1);
    k_rsum<<<256, 256, 0, stream>>>(logit, nodes, red, mslot, part);
    k_fin<<<1, 256, 0, stream>>>(part, 256, red, sslot, 0);
    int tot = nodes * HYPERN;
    k_scale_rows<<<(tot + 255) / 256, 256, 0, stream>>>(out, logit, nodes, HYPERN, red, mslot, sslot);
  };
  attn_path(uEmbeds, uHyp, u_attn, uu, USERN, 0, 1);
  attn_path(iEmbeds, iHyp, i_attn, ii, ITEMN, 2, 3);

  const int ntilesN = NNODE / 16;               // 18750
  const int disBlocks = (ntilesN + WPB - 1) / WPB;

  for (int i = 0; i < GLAYERS; ++i) {
    // --- tem = spmm(adj, x) ---
    hipMemsetAsync(tem, 0, (size_t)NNODE * LATD * 4, stream);
    k_spmm<<<(NEDGE * 16 + 255) / 256, 256, 0, stream>>>(adj_vals, adj_rows, adj_cols, x, tem, NEDGE);

    // --- hyp = [uu (uu^T x_u); ii (ii^T x_i)] ---
    {
      hipMemsetAsync(P, 0, HYPERN * LATD * 4, stream);
      int nch = (USERN + PT_CHUNK - 1) / PT_CHUNK;
      k_pt<<<(32 * nch + 7) / 8, 256, 0, stream>>>(uu, x, P, USERN);
      k_hyp<<<(USERN / 16 + 7) / 8, 256, 0, stream>>>(uu, P, hyp, USERN);
    }
    {
      hipMemsetAsync(P, 0, HYPERN * LATD * 4, stream);
      int nch = (ITEMN + PT_CHUNK - 1) / PT_CHUNK;
      k_pt<<<(32 * nch + 7) / 8, 256, 0, stream>>>(ii, x + (size_t)USERN * LATD, P, ITEMN);
      k_hyp<<<(ITEMN / 16 + 7) / 8, 256, 0, stream>>>(ii, P, hyp + (size_t)USERN * LATD, ITEMN);
    }

    // --- dis + gate per branch (in-place) ---
    for (int j = 0; j < 2; ++j) {
      float* z = (j == 0) ? tem : hyp;
      int ij = i * 2 + j;
      k_dis<<<disBlocks, 128, 0, stream>>>(
          z, NNODE,
          W1t + (size_t)(ij * 4) * 64 * 64, W2t + (size_t)(ij * 4) * 16 * 64,
          Wr1t + (size_t)ij * 64 * 32, Wr2t + (size_t)ij * 64 * 64,
          db1 + (size_t)(ij * 4) * 64, db2 + (size_t)(ij * 4) * 16,
          dbr1 + (size_t)ij * 64, dbr2 + (size_t)ij * 64,
          dwts + (size_t)ij * 4, dlng + (size_t)ij * 64, dlnb + (size_t)ij * 64);
      k_gate<<<disBlocks, 128, 0, stream>>>(
          z, x, NNODE,
          G1t + (size_t)i * 64 * 128, G2t + (size_t)i * 64 * 64,
          GR1t + (size_t)i * 64 * 128, GR2t + (size_t)i * 64 * 64,
          gbg1 + (size_t)i * 64, gbg2 + (size_t)i * 64,
          gbr1 + (size_t)i * 64, gbr2 + (size_t)i * 64,
          glng + (size_t)i * 64, glnb + (size_t)i * 64);
    }

    // --- x = gate(tem) + gate(hyp); total += x ---
    int tot = NNODE * LATD;
    k_combine<<<(tot + 255) / 256, 256, 0, stream>>>(x, total, tem, hyp, tot);
  }
}
